// HAEncoder_73048803770899
// MI455X (gfx1250) — matrix operand order repaired and checked by reading, not compiled, not run on hardware
//
#include <hip/hip_runtime.h>
#include <hip/hip_bf16.h>
#include <stdint.h>

// ---------------- problem dims ----------------
#define BB 64      // batch
#define SS 24      // sentences
#define TT 96      // words per sentence
#define EE 256     // embed dim
#define HH 512     // hidden dim
#define G4 2048    // 4*H gate width
#define KW 768     // E + H (word-LSTM K)
#define KS 1024    // H + H (sentence-LSTM K)
#define ROWS 16    // batch rows per workgroup
#define NTH 512    // threads per block (16 waves of 32)

typedef __attribute__((ext_vector_type(16))) __bf16 v16bf;
typedef __attribute__((ext_vector_type(8)))  float  v8f;

union FragU { uint32_t u[8]; v16bf v; };
union Pk8   { uint4 q; __bf16 h[8]; };   // 8 packed bf16 <-> one 16B load

__device__ __forceinline__ v8f wmma_bf16(v16bf a, v16bf b, v8f c) {
  // (neg_a, A, neg_b, B, c_mod, C, reuse_a, reuse_b)
  return __builtin_amdgcn_wmma_f32_16x16x32_bf16(false, a, false, b, (short)0, c,
                                                 false, false);
}

// A fragment: 16x32 bf16, row-major source with leading dim ld (elements).
// ISA layout: lanes 0-15 -> M=lane, K {0..7,16..23}; lanes 16-31 -> K {8..15,24..31}
__device__ __forceinline__ v16bf load_a_frag(const __bf16* base, int ld, int kb, int lane) {
  FragU f;
  const int row   = lane & 15;
  const int khalf = (lane & 16) ? 8 : 0;
  const __bf16* p = base + row * ld + kb * 32 + khalf;
#pragma unroll
  for (int j = 0; j < 8; ++j) {
    const int k = ((j < 4) ? 0 : 16) + (j & 3) * 2;
    f.u[j] = *reinterpret_cast<const uint32_t*>(p + k);
  }
  return f.v;
}

// B fragment: 32x16 bf16 where B[k][n] = W[n][k], W row-major [N][K] bf16.
// ISA layout: lanes 0-15 -> N=lane, K {0..7,16..23}; lanes 16-31 -> K {8..15,24..31} (2 K per VGPR)
__device__ __forceinline__ v16bf load_b_frag(const __bf16* W, int ld, int n0, int kb, int lane) {
  FragU f;
  const int n = n0 + (lane & 15);
  const __bf16* p = W + (size_t)n * ld + kb * 32 + ((lane & 16) ? 8 : 0);
#pragma unroll
  for (int j = 0; j < 8; ++j)
    f.u[j] = *reinterpret_cast<const uint32_t*>(p + ((j < 4) ? 0 : 16) + (j & 3) * 2);
  return f.v;
}

__device__ __forceinline__ float sigm(float x) { return 1.0f / (1.0f + __expf(-x)); }

// ---------------- LDS layout (bytes) ----------------
#define L_XHW   0        // bf16 [16][768]  : x(0..255) | word h(256..767)
#define L_CW    24576    // f32  [16][512]  : word c state
#define L_CS    57344    // f32  [16][512]  : sentence c state
#define L_HSENT 90112    // bf16 [16][512]  : sentence h state
#define L_ZW    106496   // f32  [96][16]   : word scores / alphas
#define L_ZS    112640   // f32  [24][16]   : sentence scores / alphas
#define L_XS    114176   // bf16 [16][1024] : sentence GEMM staging (aliases KA)
#define L_KA    114176   // bf16 [16][512]  : masked word h (attention A)
#define L_TOT   146944

// ---------------- workspace layout (bytes) ----------------
enum : size_t {
  OFF_WW   = 0,                                       // bf16 [2048][768]
  OFF_WS   = OFF_WW  + (size_t)G4 * KW * 2,           // bf16 [2048][1024]
  OFF_WKA  = OFF_WS  + (size_t)G4 * KS * 2,           // bf16 [512][512]
  OFF_WKS  = OFF_WKA + (size_t)HH * HH * 2,           // bf16 [512][512]
  OFF_BW   = OFF_WKS + (size_t)HH * HH * 2,           // f32  [2048]
  OFF_BS   = OFF_BW  + (size_t)G4 * 4,                // f32  [2048]
  OFF_KBUF = OFF_BS  + (size_t)G4 * 4,                // bf16 4wg x [96][16][512]
  OFF_SBUF = OFF_KBUF + (size_t)4 * TT * ROWS * HH * 2// bf16 4wg x [24][16][512]
};

// ---------------- weight prep kernels ----------------
__global__ void k_cvt_concat(__bf16* dst, const float* a, int wa,
                             const float* b, int wb, int rows) {
  const int w = wa + wb;
  const long long total = (long long)rows * w;
  const long long idx = (long long)blockIdx.x * blockDim.x + threadIdx.x;
  if (idx >= total) return;
  const int r = (int)(idx / w);
  const int c = (int)(idx % w);
  const float v = (c < wa) ? a[(size_t)r * wa + c] : b[(size_t)r * wb + (c - wa)];
  dst[idx] = (__bf16)v;
}

__global__ void k_addbias(float* dst, const float* a, const float* b, int n) {
  const int i = blockIdx.x * blockDim.x + threadIdx.x;
  if (i < n) dst[i] = a[i] + b[i];
}

// ---------------- persistent hierarchical encoder ----------------
__global__ __launch_bounds__(NTH) void hae_main(
    const int* __restrict__ captions, const unsigned char* __restrict__ umask,
    const float* __restrict__ embed,
    const float* __restrict__ waWv, const float* __restrict__ saWv,
    const __bf16* __restrict__ Ww, const __bf16* __restrict__ Ws,
    const __bf16* __restrict__ Wka, const __bf16* __restrict__ Wks,
    const float* __restrict__ bw, const float* __restrict__ bs,
    __bf16* __restrict__ kbuf, __bf16* __restrict__ sbuf,
    float* __restrict__ out)
{
  __shared__ __align__(16) unsigned char smem[L_TOT];
  __bf16* xhw   = (__bf16*)(smem + L_XHW);
  float*  cw    = (float* )(smem + L_CW);
  float*  cs    = (float* )(smem + L_CS);
  __bf16* hsent = (__bf16*)(smem + L_HSENT);
  float*  zw    = (float* )(smem + L_ZW);
  float*  zs    = (float* )(smem + L_ZS);
  __bf16* xs    = (__bf16*)(smem + L_XS);
  __bf16* ka    = (__bf16*)(smem + L_KA);

  const int tid  = threadIdx.x;
  const int wave = tid >> 5;
  const int lane = tid & 31;
  const int b0   = blockIdx.x * ROWS;
  kbuf += (size_t)blockIdx.x * TT * ROWS * HH;
  sbuf += (size_t)blockIdx.x * SS * ROWS * HH;

  // zero-init recurrent state
  for (int i = tid; i < ROWS * HH; i += NTH) {
    cw[i] = 0.f; cs[i] = 0.f; hsent[i] = (__bf16)0.f;
    xhw[(i / HH) * KW + EE + (i % HH)] = (__bf16)0.f;
  }
  for (int i = tid; i < SS * 16; i += NTH) zs[i] = 0.f;
  __syncthreads();

  const int mhi = (lane & 16) ? 8 : 0;

  for (int s = 0; s < SS; ++s) {
    for (int t = 0; t < TT; ++t) {
      // ---- stage embeddings (f32 gather -> bf16 LDS), zero word score ----
      {
        const int m = tid >> 5;
        const int c = (tid & 31) * 8;
        const int tok = captions[((b0 + m) * SS + s) * TT + t];
        const float* e = embed + (size_t)tok * EE + c;
#pragma unroll
        for (int j = 0; j < 8; ++j) xhw[m * KW + c + j] = (__bf16)e[j];
        if (tid < 16) zw[t * 16 + tid] = 0.f;
        // prefetch next word's embedding row into cache hierarchy
        if (t + 1 < TT) {
          const int tok2 = captions[((b0 + m) * SS + s) * TT + t + 1];
          __builtin_prefetch(embed + (size_t)tok2 * EE + c, 0, 3);
        }
      }
      __syncthreads();

      // ---- gate GEMM: [x|h](16xKW) @ Ww^T -> 16x2048 ; wave owns 2 gate quads.
      //      All 8 B fragments hoisted per K-block so L2 latency overlaps WMMA.
      v8f acc[8];
#pragma unroll
      for (int i = 0; i < 8; ++i) { v8f z = {}; acc[i] = z; }
      for (int kb = 0; kb < KW / 32; ++kb) {
        v16bf a = load_a_frag(xhw, KW, kb, lane);
        v16bf bf8[8];
#pragma unroll
        for (int i = 0; i < 8; ++i) {
          const int q = i >> 2, g = i & 3;
          const int n0 = g * HH + (wave * 2 + q) * 16;
          bf8[i] = load_b_frag(Ww, KW, n0, kb, lane);
        }
#pragma unroll
        for (int i = 0; i < 8; ++i)
          acc[i] = wmma_bf16(a, bf8[i], acc[i]);
      }
      __syncthreads();  // all GEMM reads of xhw complete before h update

      // ---- LSTM elementwise with mask (in accumulator registers) ----
#pragma unroll
      for (int q = 0; q < 2; ++q) {
        const int n = (wave * 2 + q) * 16 + (lane & 15);
        const float bi = bw[n], bf_ = bw[HH + n], bg = bw[2 * HH + n], bo = bw[3 * HH + n];
#pragma unroll
        for (int r = 0; r < 8; ++r) {
          const int m = r + mhi;
          const float gi = acc[q * 4 + 0][r] + bi;
          const float gf = acc[q * 4 + 1][r] + bf_;
          const float gg = acc[q * 4 + 2][r] + bg;
          const float go = acc[q * 4 + 3][r] + bo;
          const float cold = cw[m * HH + n];
          const float cn = sigm(gf) * cold + sigm(gi) * tanhf(gg);
          const float hn = sigm(go) * tanhf(cn);
          const bool mk = umask[((b0 + m) * SS + s) * TT + t] != 0;
          const float hold = (float)xhw[m * KW + EE + n];
          cw[m * HH + n] = mk ? cn : cold;
          xhw[m * KW + EE + n] = (__bf16)(mk ? hn : hold);
          const float kv = mk ? hn : 0.f;
          ka[m * HH + n] = (__bf16)kv;
          kbuf[((size_t)t * ROWS + m) * HH + n] = (__bf16)kv;
        }
      }
      __syncthreads();

      // ---- word attention score: z[t] += tanh(k @ Wka^T) . waWv ----
      {
        const int n00 = (wave * 2 + 0) * 16;
        const int n01 = (wave * 2 + 1) * 16;
        v8f za0 = {}, za1 = {};
        for (int kb = 0; kb < HH / 32; ++kb) {
          v16bf a  = load_a_frag(ka, HH, kb, lane);
          v16bf b0 = load_b_frag(Wka, HH, n00, kb, lane);
          v16bf b1 = load_b_frag(Wka, HH, n01, kb, lane);
          za0 = wmma_bf16(a, b0, za0);
          za1 = wmma_bf16(a, b1, za1);
        }
        const float wv0 = waWv[n00 + (lane & 15)];
        const float wv1 = waWv[n01 + (lane & 15)];
#pragma unroll
        for (int r = 0; r < 8; ++r) {
          atomicAdd(&zw[t * 16 + r + mhi], tanhf(za0[r]) * wv0);
          atomicAdd(&zw[t * 16 + r + mhi], tanhf(za1[r]) * wv1);
        }
      }
      __syncthreads();
    } // t

    // ---- word softmax over T (per batch row), in place ----
    if (tid < 16) {
      const int m = tid;
      float mx = -1e30f;
      for (int t = 0; t < TT; ++t) mx = fmaxf(mx, zw[t * 16 + m]);
      float sum = 0.f;
      for (int t = 0; t < TT; ++t) { const float e = __expf(zw[t * 16 + m] - mx); zw[t * 16 + m] = e; sum += e; }
      const float inv = 1.f / sum;
      for (int t = 0; t < TT; ++t) zw[t * 16 + m] *= inv;
    }
    __syncthreads();

    // ---- ctx = sum_t alpha_t * k_t ; stage xs = [ctx | hsent] ----
    {
      const int m = tid >> 5;
      const int cb = (tid & 31) * 16;
      float av[16];
#pragma unroll
      for (int j = 0; j < 16; ++j) av[j] = 0.f;
      for (int t = 0; t < TT; ++t) {
        const float al = zw[t * 16 + m];
        const __bf16* kp = kbuf + ((size_t)t * ROWS + m) * HH + cb;
        Pk8 p0, p1;
        p0.q = *reinterpret_cast<const uint4*>(kp);
        p1.q = *reinterpret_cast<const uint4*>(kp + 8);
#pragma unroll
        for (int j = 0; j < 8; ++j) {
          av[j]     += al * (float)p0.h[j];
          av[j + 8] += al * (float)p1.h[j];
        }
      }
#pragma unroll
      for (int j = 0; j < 16; ++j) {
        xs[m * KS + cb + j] = (__bf16)av[j];
        xs[m * KS + HH + cb + j] = hsent[m * HH + cb + j];
      }
    }
    __syncthreads();

    // ---- sentence LSTM GEMM: [ctx|sh](16x1024) @ Ws^T ----
    v8f sacc[8];
#pragma unroll
    for (int i = 0; i < 8; ++i) { v8f z = {}; sacc[i] = z; }
    for (int kb = 0; kb < KS / 32; ++kb) {
      v16bf a = load_a_frag(xs, KS, kb, lane);
      v16bf bf8[8];
#pragma unroll
      for (int i = 0; i < 8; ++i) {
        const int q = i >> 2, g = i & 3;
        const int n0 = g * HH + (wave * 2 + q) * 16;
        bf8[i] = load_b_frag(Ws, KS, n0, kb, lane);
      }
#pragma unroll
      for (int i = 0; i < 8; ++i)
        sacc[i] = wmma_bf16(a, bf8[i], sacc[i]);
    }
    __syncthreads();

    // ---- sentence LSTM elementwise (no mask) ----
#pragma unroll
    for (int q = 0; q < 2; ++q) {
      const int n = (wave * 2 + q) * 16 + (lane & 15);
      const float bi = bs[n], bf_ = bs[HH + n], bg = bs[2 * HH + n], bo = bs[3 * HH + n];
#pragma unroll
      for (int r = 0; r < 8; ++r) {
        const int m = r + mhi;
        const float gi = sacc[q * 4 + 0][r] + bi;
        const float gf = sacc[q * 4 + 1][r] + bf_;
        const float gg = sacc[q * 4 + 2][r] + bg;
        const float go = sacc[q * 4 + 3][r] + bo;
        const float cold = cs[m * HH + n];
        const float cn = sigm(gf) * cold + sigm(gi) * tanhf(gg);
        const float hn = sigm(go) * tanhf(cn);
        cs[m * HH + n] = cn;
        hsent[m * HH + n] = (__bf16)hn;
        sbuf[((size_t)s * ROWS + m) * HH + n] = (__bf16)hn;
      }
    }
    __syncthreads();

    // ---- sentence attention score: z[s] += tanh(sh @ Wks^T) . saWv ----
    {
      const int n00 = (wave * 2 + 0) * 16;
      const int n01 = (wave * 2 + 1) * 16;
      v8f za0 = {}, za1 = {};
      for (int kb = 0; kb < HH / 32; ++kb) {
        v16bf a  = load_a_frag(hsent, HH, kb, lane);
        v16bf b0 = load_b_frag(Wks, HH, n00, kb, lane);
        v16bf b1 = load_b_frag(Wks, HH, n01, kb, lane);
        za0 = wmma_bf16(a, b0, za0);
        za1 = wmma_bf16(a, b1, za1);
      }
      const float wv0 = saWv[n00 + (lane & 15)];
      const float wv1 = saWv[n01 + (lane & 15)];
#pragma unroll
      for (int r = 0; r < 8; ++r) {
        atomicAdd(&zs[s * 16 + r + mhi], tanhf(za0[r]) * wv0);
        atomicAdd(&zs[s * 16 + r + mhi], tanhf(za1[r]) * wv1);
      }
    }
    __syncthreads();
  } // s

  // ---- sentence softmax -> alpha output ----
  if (tid < 16) {
    const int m = tid;
    float mx = -1e30f;
    for (int s = 0; s < SS; ++s) mx = fmaxf(mx, zs[s * 16 + m]);
    float sum = 0.f;
    for (int s = 0; s < SS; ++s) { const float e = __expf(zs[s * 16 + m] - mx); zs[s * 16 + m] = e; sum += e; }
    const float inv = 1.f / sum;
    for (int s = 0; s < SS; ++s) {
      const float a = zs[s * 16 + m] * inv;
      zs[s * 16 + m] = a;
      out[(size_t)BB * HH + (size_t)(b0 + m) * SS + s] = a;
    }
  }
  __syncthreads();

  // ---- final context = sum_s alpha_s * sh_s ----
  {
    const int m = tid >> 5;
    const int cb = (tid & 31) * 16;
    float av[16];
#pragma unroll
    for (int j = 0; j < 16; ++j) av[j] = 0.f;
    for (int s = 0; s < SS; ++s) {
      const float al = zs[s * 16 + m];
      const __bf16* kp = sbuf + ((size_t)s * ROWS + m) * HH + cb;
      Pk8 p0, p1;
      p0.q = *reinterpret_cast<const uint4*>(kp);
      p1.q = *reinterpret_cast<const uint4*>(kp + 8);
#pragma unroll
      for (int j = 0; j < 8; ++j) {
        av[j]     += al * (float)p0.h[j];
        av[j + 8] += al * (float)p1.h[j];
      }
    }
#pragma unroll
    for (int j = 0; j < 16; ++j) out[(size_t)(b0 + m) * HH + cb + j] = av[j];
  }
}

extern "C" void kernel_launch(void* const* d_in, const int* in_sizes, int n_in,
                              void* d_out, int out_size, void* d_ws, size_t ws_size,
                              hipStream_t stream) {
  const int*           captions = (const int*)d_in[0];
  const unsigned char* umask    = (const unsigned char*)d_in[1];
  const float* embed = (const float*)d_in[2];
  const float* w_Wih = (const float*)d_in[3];
  const float* w_Whh = (const float*)d_in[4];
  const float* w_bih = (const float*)d_in[5];
  const float* w_bhh = (const float*)d_in[6];
  const float* s_Wih = (const float*)d_in[7];
  const float* s_Whh = (const float*)d_in[8];
  const float* s_bih = (const float*)d_in[9];
  const float* s_bhh = (const float*)d_in[10];
  const float* wa_Wk = (const float*)d_in[11];
  const float* wa_Wv = (const float*)d_in[12];
  const float* sa_Wk = (const float*)d_in[13];
  const float* sa_Wv = (const float*)d_in[14];

  char* ws = (char*)d_ws;
  __bf16* Ww   = (__bf16*)(ws + OFF_WW);
  __bf16* Wsn  = (__bf16*)(ws + OFF_WS);
  __bf16* Wka  = (__bf16*)(ws + OFF_WKA);
  __bf16* Wks  = (__bf16*)(ws + OFF_WKS);
  float*  bw   = (float*)(ws + OFF_BW);
  float*  bs   = (float*)(ws + OFF_BS);
  __bf16* kbuf = (__bf16*)(ws + OFF_KBUF);
  __bf16* sbuf = (__bf16*)(ws + OFF_SBUF);
  float*  out  = (float*)d_out;

  // weight prep: f32 -> bf16, concat [Wih|Whh]; fuse biases
  { const int tot = G4 * KW; k_cvt_concat<<<(tot + 255) / 256, 256, 0, stream>>>(Ww,  w_Wih, EE, w_Whh, HH, G4); }
  { const int tot = G4 * KS; k_cvt_concat<<<(tot + 255) / 256, 256, 0, stream>>>(Wsn, s_Wih, HH, s_Whh, HH, G4); }
  { const int tot = HH * HH; k_cvt_concat<<<(tot + 255) / 256, 256, 0, stream>>>(Wka, wa_Wk, HH, wa_Wk, 0, HH); }
  { const int tot = HH * HH; k_cvt_concat<<<(tot + 255) / 256, 256, 0, stream>>>(Wks, sa_Wk, HH, sa_Wk, 0, HH); }
  k_addbias<<<(G4 + 255) / 256, 256, 0, stream>>>(bw, w_bih, w_bhh, G4);
  k_addbias<<<(G4 + 255) / 256, 256, 0, stream>>>(bs, s_bih, s_bhh, G4);

  // persistent encoder: 4 workgroups x 16 batch rows, 16 waves each
  hae_main<<<BB / ROWS, NTH, 0, stream>>>(captions, umask, embed, wa_Wv, sa_Wv,
                                          Ww, Wsn, Wka, Wks, bw, bs, kbuf, sbuf, out);
  (void)in_sizes; (void)n_in; (void)out_size; (void)ws_size;
}
